// ViTVectorQuantizer_56856777064890
// MI455X (gfx1250) — compile-verified, matrix operand order checked by reading, not run on hardware
//
#include <hip/hip_runtime.h>
#include <hip/hip_bf16.h>
#include <math.h>

typedef __attribute__((ext_vector_type(16))) _Float16 v16h;
typedef __attribute__((ext_vector_type(8)))  _Float16 v8h;
typedef __attribute__((ext_vector_type(8)))  float    v8f;

#define EMBED_D 256
#define STAGE_COLS 32          // codebook rows staged per double-buffer slot
#define ROW_H 264              // 256 halves + 8 halves (16B) anti-bank-conflict pad

// Scheduling-group pins (no-op if builtin unavailable).
// Masks: 0x0008 = MFMA/WMMA, 0x0100 = DS read.
#if __has_builtin(__builtin_amdgcn_sched_group_barrier)
#define SGB(mask, size, id) __builtin_amdgcn_sched_group_barrier((mask), (size), (id))
#else
#define SGB(mask, size, id)
#endif

// ---------------------------------------------------------------------------
// Pass 1/2: L2-normalize rows of a [nrows, 256] f32 matrix.
// One wave per row: 32 lanes x 8 contiguous floats. Writes f16 (for WMMA)
// and optionally f32 (normalized codebook needed for the gather output).
// ---------------------------------------------------------------------------
__global__ __launch_bounds__(256) void vq_normalize_rows(
    const float* __restrict__ src, _Float16* __restrict__ dstH,
    float* __restrict__ dstF, int nrows) {
  int wave = (int)((blockIdx.x * blockDim.x + threadIdx.x) >> 5);
  int lane = threadIdx.x & 31;
  if (wave >= nrows) return;
  const float* row = src + (size_t)wave * EMBED_D;
  const float4* row4 = (const float4*)row;
  float4 a = row4[lane * 2 + 0];
  float4 b = row4[lane * 2 + 1];
  float s = a.x * a.x + a.y * a.y + a.z * a.z + a.w * a.w +
            b.x * b.x + b.y * b.y + b.z * b.z + b.w * b.w;
#pragma unroll
  for (int m = 1; m <= 16; m <<= 1) s += __shfl_xor(s, m);
  float inv = 1.0f / fmaxf(sqrtf(s), 1e-12f);

  float v[8] = {a.x * inv, a.y * inv, a.z * inv, a.w * inv,
                b.x * inv, b.y * inv, b.z * inv, b.w * inv};
  size_t base = (size_t)wave * EMBED_D + (size_t)lane * 8;
#pragma unroll
  for (int i = 0; i < 8; ++i) dstH[base + i] = (_Float16)v[i];
  if (dstF) {
#pragma unroll
    for (int i = 0; i < 8; ++i) dstF[base + i] = v[i];
  }
}

// ---------------------------------------------------------------------------
// Pass 3: fused GEMM (X_norm @ E_norm^T) + row-wise argmax via WMMA.
//
// Block = 256 threads = 8 waves = 128 X-rows. The codebook tile is staged in
// LDS ONCE per block (8x L2-traffic reduction vs per-wave streaming) with
// double buffering: the cooperative fill of buffer `next` is issued before
// the WMMA work on buffer `cur`, one barrier per 32-column iteration.
//
// The 16 WMMAs of a stage (two interleaved accumulator chains) are pinned
// into a software pipeline with sched_group_barrier: the first two K-chunks'
// DS reads are hoisted ahead, then groups of {2 WMMA, 4 DS-read} alternate so
// chunk c+1's LDS loads are in flight behind chunk c's WMMAs (partial dscnt
// waits instead of a full drain before every WMMA).
//
// LDS rows padded to 132 dwords (== 4 mod 64 banks): the 16 lanes of a
// fragment read hit disjoint bank groups.
// C layout: lane l holds column n = l&15; VGPR r holds row m = r + 8*(l>>4).
// ---------------------------------------------------------------------------
__global__ __launch_bounds__(256) void vq_argmax_wmma(
    const _Float16* __restrict__ xh, const _Float16* __restrict__ eh,
    int* __restrict__ idxOut, float* __restrict__ idxOutF, int vocab) {
  __shared__ _Float16 stage[2][STAGE_COLS * ROW_H];

  const int tid  = threadIdx.x;
  const int lane = tid & 31;
  const int wave = tid >> 5;
  const int rowA = (int)blockIdx.x * 128 + wave * 16;
  const int m  = lane & 15;   // A row / C column selector
  const int kh = lane >> 4;   // K-half within 32-wide chunk

  // Cooperative fill mapping: 32 rows x 512B; each thread moves 64B.
  const int frow = tid >> 3;  // 0..31: stage row
  const int fseg = tid & 7;   // 0..7 : 64B segment within row
  auto fill = [&](int buf, int tcol) {
    const uint4* src =
        (const uint4*)(eh + (size_t)(tcol + frow) * EMBED_D) + fseg * 4;
    uint4* dst = (uint4*)(&stage[buf][frow * ROW_H]) + fseg * 4;
    uint4 a0 = src[0], a1 = src[1], a2 = src[2], a3 = src[3];
    dst[0] = a0; dst[1] = a1; dst[2] = a2; dst[3] = a3;
  };

  // A fragments: 16 rows x 256 K, resident for the whole scan (8 x v16h).
  const _Float16* abase = xh + (size_t)(rowA + m) * EMBED_D + kh * 16;
  v16h afrag[8];
#pragma unroll
  for (int c = 0; c < 8; ++c)
    afrag[c] = *(const v16h*)(abase + c * 32);

  float bestVal[8];
  int   bestIdx[8];
#pragma unroll
  for (int r = 0; r < 8; ++r) { bestVal[r] = -3.0e38f; bestIdx[r] = 0; }

  fill(0, 0);
  __syncthreads();

  for (int t = 0; t < vocab; t += STAGE_COLS) {
    const int cur = (t >> 5) & 1;
    if (t + STAGE_COLS < vocab) fill(cur ^ 1, t + STAGE_COLS);  // uniform

    // Base LDS offsets for the two tiles' fragments of this wave's lanes.
    const _Float16* p0 = &stage[cur][(m)      * ROW_H + kh * 16];
    const _Float16* p1 = &stage[cur][(16 + m) * ROW_H + kh * 16];

    v8f acc0 = {};
    v8f acc1 = {};
#pragma unroll
    for (int c = 0; c < 8; ++c) {
      union { v16h v; v8h h[2]; } u0, u1;
      u0.h[0] = *(const v8h*)(p0 + c * 32);
      u0.h[1] = *(const v8h*)(p0 + c * 32 + 8);
      u1.h[0] = *(const v8h*)(p1 + c * 32);
      u1.h[1] = *(const v8h*)(p1 + c * 32 + 8);
      acc0 = __builtin_amdgcn_wmma_f32_16x16x32_f16(
          false, afrag[c], false, u0.v, (short)0, acc0, false, false);
      acc1 = __builtin_amdgcn_wmma_f32_16x16x32_f16(
          false, afrag[c], false, u1.v, (short)0, acc1, false, false);
    }

    // Pin a 1-chunk-deep pipeline: hoist two chunks of DS reads, then
    // alternate {2 WMMA, 4 DS-read} so loads run ahead of the matrix ops.
    SGB(0x0100, 8, 0);            // chunk 0 + chunk 1 fragment loads
#pragma unroll
    for (int c = 0; c < 7; ++c) {
      SGB(0x0008, 2, 0);          // chunk c: wmma acc0, wmma acc1
      SGB(0x0100, 4, 0);          // chunk c+2 fragment loads
    }
    SGB(0x0008, 2, 0);            // final chunk's wmmas

    const int col0 = t + m;
    const int col1 = t + 16 + m;
#pragma unroll
    for (int r = 0; r < 8; ++r) {
      float v0 = acc0[r];
      float v1 = acc1[r];
      // Indices strictly increase within a lane -> strict > keeps first-min.
      if (v0 > bestVal[r]) { bestVal[r] = v0; bestIdx[r] = col0; }
      if (v1 > bestVal[r]) { bestVal[r] = v1; bestIdx[r] = col1; }
    }
    __syncthreads();
  }

  // Reduce across the 16 lanes sharing the same 8 rows (masks stay in-half).
#pragma unroll
  for (int msk = 1; msk <= 8; msk <<= 1) {
#pragma unroll
    for (int r = 0; r < 8; ++r) {
      float ov = __shfl_xor(bestVal[r], msk);
      int   oi = __shfl_xor(bestIdx[r], msk);
      if (ov > bestVal[r] || (ov == bestVal[r] && oi < bestIdx[r])) {
        bestVal[r] = ov; bestIdx[r] = oi;
      }
    }
  }

  if (m == 0) {  // lane 0 -> rows 0..7, lane 16 -> rows 8..15
#pragma unroll
    for (int r = 0; r < 8; ++r) {
      int row = rowA + r + 8 * kh;
      idxOut[row]  = bestIdx[r];
      idxOutF[row] = (float)bestIdx[r];
    }
  }
}

// ---------------------------------------------------------------------------
// Pass 4: gather quantized rows (= normalized codebook rows) into d_out and
// accumulate the squared error as deterministic per-block partials.
// One block (256 threads) per token; dim == threadIdx.x.
// ---------------------------------------------------------------------------
__global__ __launch_bounds__(256) void vq_gather_sse(
    const float* __restrict__ x, const float* __restrict__ enorm,
    const int* __restrict__ idx, float* __restrict__ qout,
    float* __restrict__ partial) {
  __shared__ float red[256];
  int tok = blockIdx.x;
  int d   = threadIdx.x;
  int code = idx[tok];
  size_t g = (size_t)tok * EMBED_D + d;
  float q = enorm[(size_t)code * EMBED_D + d];
  qout[g] = q;  // straight-through output == quantized value
  float diff = q - x[g];
  red[d] = diff * diff;
  __syncthreads();
#pragma unroll
  for (int s = 128; s > 0; s >>= 1) {
    if (d < s) red[d] += red[d + s];
    __syncthreads();
  }
  if (d == 0) partial[tok] = red[0];
}

// ---------------------------------------------------------------------------
// Pass 5: integer histogram of code usage (order-independent -> deterministic)
// ---------------------------------------------------------------------------
__global__ __launch_bounds__(256) void vq_hist(
    const int* __restrict__ idx, int* __restrict__ counts, int n) {
  int i = (int)(blockIdx.x * blockDim.x + threadIdx.x);
  if (i < n) atomicAdd(&counts[idx[i]], 1);
}

// ---------------------------------------------------------------------------
// Pass 6: single-block finalize (fixed-order sums -> deterministic).
// loss = 1.25 * SSE / (N*D);  perplexity = exp(-sum p*log(p+1e-5)).
// ---------------------------------------------------------------------------
__global__ __launch_bounds__(256) void vq_finalize(
    const float* __restrict__ partial, int nPartials,
    const int* __restrict__ counts, int vocab, int nTokens,
    float* __restrict__ out, float lossScale) {
  __shared__ float r1[256];
  __shared__ float r2[256];
  int t = threadIdx.x;
  float a = 0.0f;
  for (int i = t; i < nPartials; i += 256) a += partial[i];
  float invN = 1.0f / (float)nTokens;
  float b = 0.0f;
  for (int v = t; v < vocab; v += 256) {
    float p = (float)counts[v] * invN;
    b += p * logf(p + 1e-5f);
  }
  r1[t] = a; r2[t] = b;
  __syncthreads();
#pragma unroll
  for (int s = 128; s > 0; s >>= 1) {
    if (t < s) { r1[t] += r1[t + s]; r2[t] += r2[t + s]; }
    __syncthreads();
  }
  if (t == 0) {
    out[0] = r1[0] * lossScale;   // q_latent + 0.25*e_latent = 1.25*MSE
    out[1] = expf(-r2[0]);        // perplexity
  }
}

// ---------------------------------------------------------------------------
extern "C" void kernel_launch(void* const* d_in, const int* in_sizes, int n_in,
                              void* d_out, int out_size, void* d_ws, size_t ws_size,
                              hipStream_t stream) {
  const float* x   = (const float*)d_in[0];   // [32,1024,256] f32
  const float* emb = (const float*)d_in[1];   // [8192,256]    f32
  const int N = in_sizes[0] / EMBED_D;        // 32768 tokens
  const int V = in_sizes[1] / EMBED_D;        // 8192 codes

  // Workspace layout (all regions naturally aligned; ~28.4 MB total).
  char* ws = (char*)d_ws;
  float*    enF   = (float*)ws;                                       // V*256 f32
  _Float16* eH    = (_Float16*)(ws + (size_t)V * EMBED_D * 4);        // V*256 f16
  _Float16* xH    = (_Float16*)((char*)eH + (size_t)V * EMBED_D * 2); // N*256 f16
  int*      idx   = (int*)((char*)xH + (size_t)N * EMBED_D * 2);      // N  i32
  int*      cnt   = (int*)((char*)idx + (size_t)N * 4);               // V  i32
  float*    part  = (float*)((char*)cnt + (size_t)V * 4);             // N  f32

  float* qout    = (float*)d_out;                 // [N*256] quantized_st
  float* scalars = qout + (size_t)N * EMBED_D;    // [0]=loss, [1]=perplexity
  float* idxF    = scalars + 2;                   // [N] indices as f32

  hipMemsetAsync(cnt, 0, (size_t)V * sizeof(int), stream);

  vq_normalize_rows<<<V / 8, 256, 0, stream>>>(emb, eH, enF, V);
  vq_normalize_rows<<<N / 8, 256, 0, stream>>>(x, xH, nullptr, N);
  vq_argmax_wmma<<<N / 128, 256, 0, stream>>>(xH, eH, idx, idxF, V);
  vq_gather_sse<<<N, 256, 0, stream>>>(x, enF, idx, qout, part);
  vq_hist<<<(N + 255) / 256, 256, 0, stream>>>(idx, cnt, N);
  vq_finalize<<<1, 256, 0, stream>>>(part, N, cnt, V, N, scalars,
                                     1.25f / ((float)N * (float)EMBED_D));
}
